// Model_TCN_SNN_6837587935377
// MI455X (gfx1250) — matrix operand order repaired and checked by reading, not compiled, not run on hardware
//
#include <hip/hip_runtime.h>
#include <hip/hip_bf16.h>
#include <math.h>

typedef __bf16 bf16;
typedef __attribute__((ext_vector_type(16))) __bf16 v16bf;
typedef __attribute__((ext_vector_type(8)))  __bf16 v8bf;
typedef __attribute__((ext_vector_type(8)))  float  v8f;

#define BSZ   32
#define TLEN  4096
#define HID   192
#define EPS   1e-5f

// ---- CDNA5 async global->LDS path (guarded; falls back to sync loads) ------
#if defined(__has_builtin)
#if __has_builtin(__builtin_amdgcn_global_load_async_to_lds_b32)
#define HAS_ASYNC_LDS 1
#endif
#endif

#if HAS_ASYNC_LDS
typedef int __attribute__((address_space(1))) gint;
typedef int __attribute__((address_space(3))) lint;
__device__ __forceinline__ void async_g2l_b32(const float* g, float* l) {
  __builtin_amdgcn_global_load_async_to_lds_b32(
      (gint*)(void*)const_cast<float*>(g), (lint*)(void*)l, 0, 0);
}
__device__ __forceinline__ void wait_asynccnt0() {
#if __has_builtin(__builtin_amdgcn_s_wait_asynccnt)
  __builtin_amdgcn_s_wait_asynccnt(0);
#else
  asm volatile("s_wait_asynccnt 0x0" ::: "memory");
#endif
}
#endif

// ---------------------------------------------------------------------------
// Weight packing: unified 8-tap layout [d][kchunk32][m][32] (bf16), taps d=0..6
// cover K3 (d=2..4), K5 (d=1..5), K7 (d=0..6); d=7 = residual 1x1.
// Also folds conv bias + BN into per-channel scale/shift, residual bias.
// ---------------------------------------------------------------------------
__global__ __launch_bounds__(256) void pack_tcn(
    const float* __restrict__ w3, const float* __restrict__ b3,
    const float* __restrict__ w5, const float* __restrict__ b5,
    const float* __restrict__ w7, const float* __restrict__ b7,
    const float* __restrict__ rw, const float* __restrict__ rb,
    const float* __restrict__ bng, const float* __restrict__ bnb,
    const float* __restrict__ bnm, const float* __restrict__ bnv,
    bf16* __restrict__ Wp, float* __restrict__ prm, int Cin, int Cpad)
{
  int i = blockIdx.x * blockDim.x + threadIdx.x;
  if (i < HID) {
    float inv = bng[i] * rsqrtf(bnv[i] + EPS);
    int g = i >> 6, mo = i & 63;
    float bias = (g == 0) ? b3[mo] : (g == 1) ? b5[mo] : b7[mo];
    prm[i]         = inv;                            // scale
    prm[HID + i]   = (bias - bnm[i]) * inv + bnb[i]; // shift (bias folded)
    prm[2*HID + i] = rb[i];                          // residual bias
  }
  const int KC = Cpad >> 5;
  const int total = 8 * HID * Cpad;
  for (; i < total; i += gridDim.x * blockDim.x) {
    int d   = i / (HID * Cpad);
    int rem = i - d * HID * Cpad;
    int m   = rem / Cpad;
    int k   = rem - m * Cpad;
    float v = 0.f;
    if (k < Cin) {
      if (d == 7) {
        v = rw[m * Cin + k];
      } else {
        int g = m >> 6, mo = m & 63;
        if (g == 0)      { int wk = d - 2; if (wk >= 0 && wk < 3) v = w3[(mo * Cin + k) * 3 + wk]; }
        else if (g == 1) { int wk = d - 1; if (wk >= 0 && wk < 5) v = w5[(mo * Cin + k) * 5 + wk]; }
        else             {                                        v = w7[(mo * Cin + k) * 7 + d ]; }
      }
    }
    int kc = k >> 5, kw = k & 31;
    Wp[(((size_t)d * KC + kc) * HID + m) * 32 + kw] = (bf16)v;
  }
}

// ---------------------------------------------------------------------------
// TCN block as implicit GEMM via v_wmma_f32_16x16x32_bf16.
// Block = one (batch, 64-time-step tile); 8 waves, each wave owns 48(M)x32(N).
// LDS holds X tile transposed [t_local][cin] so B-fragments are contiguous.
// out = relu(conv*scale + shift) + res + res_bias, stored bf16 [b][m][t].
// ---------------------------------------------------------------------------
template<typename TIN>
__global__ __launch_bounds__(256) void tcn_conv(
    const TIN* __restrict__ Xg, const bf16* __restrict__ Wp,
    const float* __restrict__ prm, bf16* __restrict__ Yg,
    int Cin, int Cpad, int dil)
{
  __shared__ bf16  Xs[20736];        // max 70*296 (layer1)
  __shared__ float scs[HID], shs[HID], rbs[HID];
  const int tid  = threadIdx.x;
  const int b    = blockIdx.x >> 6;
  const int tb   = (blockIdx.x & 63) * 64;
  const int S    = Cpad + 8;         // LDS row stride (elems), *2B is 16B-multiple
  const int halo = 3 * dil;
  const int W    = 64 + 2 * halo;

  // per-channel BN params: async DMA into LDS, overlapped with tile staging
#if HAS_ASYNC_LDS
  if (tid < HID) {
    async_g2l_b32(prm + tid,         &scs[tid]);
    async_g2l_b32(prm + HID + tid,   &shs[tid]);
    async_g2l_b32(prm + 2*HID + tid, &rbs[tid]);
  }
#else
  if (tid < HID) { scs[tid] = prm[tid]; shs[tid] = prm[HID + tid]; rbs[tid] = prm[2*HID + tid]; }
#endif

  // load + transpose X tile into LDS (coalesced along t), zero-pad halo & Cpad
  const int total = W * Cpad;
  for (int i = tid; i < total; i += 256) {
    int c  = i / W, tl = i - c * W;
    int tg = tb - halo + tl;
    float v = 0.f;
    if (c < Cin && tg >= 0 && tg < TLEN) v = (float)Xg[((size_t)b * Cin + c) * TLEN + tg];
    Xs[tl * S + c] = (bf16)v;
  }
#if HAS_ASYNC_LDS
  wait_asynccnt0();
#endif
  __syncthreads();

  const int lane  = tid & 31;
  const int wv    = tid >> 5;
  const int mbase = (wv & 3) * 48;   // 4 M-strips of 48
  const int nbase = (wv >> 2) * 32;  // 2 N-strips of 32
  const int ln15  = lane & 15;
  const int khalf = lane >> 4;
  const int KC    = Cpad >> 5;

  v8f accC[3][2] = {}, accR[3][2] = {};

  for (int k0 = 0; k0 < KC; ++k0) {
    // prefetch next K-chunk of the weight stream (global_prefetch)
    {
      const int kn = (k0 + 1 < KC) ? k0 + 1 : k0;
      const bf16* pf0 = &Wp[(((size_t)0 * KC + kn) * HID + mbase + ln15) * 32];
      const bf16* pf4 = &Wp[(((size_t)4 * KC + kn) * HID + mbase + ln15) * 32];
      __builtin_prefetch(pf0, 0, 1);
      __builtin_prefetch(pf4, 0, 1);
    }
    #pragma unroll
    for (int d = 0; d < 8; ++d) {
      const int sh = ((d < 7) ? d : 3) * dil;   // residual tap = center
      v16bf bfr[2];
      #pragma unroll
      for (int nt = 0; nt < 2; ++nt) {
        const int tl = nbase + nt * 16 + ln15 + sh;
        const bf16* p = &Xs[tl * S + k0 * 32 + khalf * 16];
        v8bf lo = *(const v8bf*)p;
        v8bf hi = *(const v8bf*)(p + 8);
        bfr[nt] = __builtin_shufflevector(lo, hi, 0,1,2,3,4,5,6,7,8,9,10,11,12,13,14,15);
      }
      #pragma unroll
      for (int mt = 0; mt < 3; ++mt) {
        const int m = mbase + mt * 16 + ln15;
        const bf16* wp = &Wp[(((size_t)d * KC + k0) * HID + m) * 32];
        v8bf alo = *(const v8bf*)(wp + khalf * 16);
        v8bf ahi = *(const v8bf*)(wp + khalf * 16 + 8);
        v16bf afr = __builtin_shufflevector(alo, ahi, 0,1,2,3,4,5,6,7,8,9,10,11,12,13,14,15);
        #pragma unroll
        for (int nt = 0; nt < 2; ++nt) {
          if (d < 7)
            accC[mt][nt] = __builtin_amdgcn_wmma_f32_16x16x32_bf16(
                false, afr, false, bfr[nt], (short)0, accC[mt][nt], false, false);
          else
            accR[mt][nt] = __builtin_amdgcn_wmma_f32_16x16x32_bf16(
                false, afr, false, bfr[nt], (short)0, accR[mt][nt], false, false);
        }
      }
    }
  }

  // epilogue: BN + ReLU + residual, store bf16 per C/D VGPR layout
  #pragma unroll
  for (int mt = 0; mt < 3; ++mt)
  #pragma unroll
  for (int nt = 0; nt < 2; ++nt) {
    const int t = tb + nbase + nt * 16 + ln15;
    #pragma unroll
    for (int r = 0; r < 8; ++r) {
      const int m = mbase + mt * 16 + khalf * 8 + r;
      float v = accC[mt][nt][r] * scs[m] + shs[m];
      v = fmaxf(v, 0.f) + accR[mt][nt][r] + rbs[m];
      Yg[((size_t)b * HID + m) * TLEN + t] = (bf16)v;
    }
  }
}

// ---------------------------------------------------------------------------
// Final channel-BN fused with parametric LIF scan (sequential in T).
// One thread per (b, c); 8-wide vector loads along T.
// ---------------------------------------------------------------------------
__global__ __launch_bounds__(HID) void lif_scan(
    const bf16* __restrict__ H, const float* __restrict__ bng,
    const float* __restrict__ bnb, const float* __restrict__ bnm,
    const float* __restrict__ bnv, const float* __restrict__ lifw,
    bf16* __restrict__ SPK)
{
  const int b = blockIdx.x, c = threadIdx.x;
  const float sc = bng[c] * rsqrtf(bnv[c] + EPS);
  const float sf = bnb[c] - bnm[c] * sc;
  const float sw = 1.f / (1.f + expf(-lifw[0]));
  const bf16* src = H   + ((size_t)b * HID + c) * TLEN;
  bf16*       dst = SPK + ((size_t)b * HID + c) * TLEN;
  float v = 0.f;
  for (int t = 0; t < TLEN; t += 8) {
    v8bf h = *(const v8bf*)(src + t);
    v8bf s;
    #pragma unroll
    for (int j = 0; j < 8; ++j) {
      float x = (float)h[j] * sc + sf;
      v += (x - v) * sw;
      float sp = (v >= 1.f) ? 1.f : 0.f;
      s[j] = (bf16)sp;
      v *= (1.f - sp);
    }
    *(v8bf*)(dst + t) = s;
  }
}

// logits[b][t] = tanh(1) * sum_c spk[b][c][t]*w[c] + b   (spikes are 0/1)
__global__ __launch_bounds__(256) void attn_logits(
    const bf16* __restrict__ SPK, const float* __restrict__ aw,
    const float* __restrict__ ab, float* __restrict__ lg)
{
  __shared__ float ws[HID];
  const int b = blockIdx.y;
  const int t = blockIdx.x * 256 + threadIdx.x;
  if (threadIdx.x < HID) ws[threadIdx.x] = aw[threadIdx.x];
  __syncthreads();
  float acc = 0.f;
  const bf16* base = SPK + (size_t)b * HID * TLEN + t;
  for (int c = 0; c < HID; ++c) acc += (float)base[(size_t)c * TLEN] * ws[c];
  lg[b * TLEN + t] = 0.761594155955765f * acc + ab[0];
}

__global__ __launch_bounds__(256) void softmax_t(
    const float* __restrict__ lg, float* __restrict__ wsoft)
{
  __shared__ float red[256];
  const int b = blockIdx.x, tid = threadIdx.x;
  const float* L = lg + b * TLEN;
  float mx = -1e30f;
  for (int t = tid; t < TLEN; t += 256) mx = fmaxf(mx, L[t]);
  red[tid] = mx; __syncthreads();
  for (int s = 128; s > 0; s >>= 1) { if (tid < s) red[tid] = fmaxf(red[tid], red[tid + s]); __syncthreads(); }
  mx = red[0]; __syncthreads();
  float sm = 0.f;
  for (int t = tid; t < TLEN; t += 256) sm += expf(L[t] - mx);
  red[tid] = sm; __syncthreads();
  for (int s = 128; s > 0; s >>= 1) { if (tid < s) red[tid] += red[tid + s]; __syncthreads(); }
  const float inv = 1.f / red[0];
  for (int t = tid; t < TLEN; t += 256) wsoft[b * TLEN + t] = expf(L[t] - mx) * inv;
}

__global__ __launch_bounds__(HID) void attn_pool(
    const bf16* __restrict__ SPK, const float* __restrict__ wsoft,
    float* __restrict__ pooled)
{
  const int b = blockIdx.x, c = threadIdx.x;
  const bf16* s = SPK + ((size_t)b * HID + c) * TLEN;
  const float* w = wsoft + b * TLEN;
  float acc = 0.f;
  for (int t = 0; t < TLEN; ++t) acc += (float)s[t] * w[t];
  pooled[b * HID + c] = acc;
}

// fc1+bn+gelu -> fc2+bn+gelu -> fc3 ; one block, all staged in LDS
__global__ __launch_bounds__(256) void classifier(
    const float* __restrict__ pooled,
    const float* fc1w, const float* fc1b, const float* g1, const float* b1, const float* m1, const float* v1,
    const float* fc2w, const float* fc2b, const float* g2, const float* b2, const float* m2, const float* v2,
    const float* fc3w, const float* fc3b, float* __restrict__ out)
{
  __shared__ float P[BSZ * HID];
  __shared__ float Z1[BSZ * 128];
  __shared__ float Z2[BSZ * 64];
  const int tid = threadIdx.x;
  for (int i = tid; i < BSZ * HID; i += 256) P[i] = pooled[i];
  __syncthreads();
  for (int i = tid; i < BSZ * 128; i += 256) {
    int r = i >> 7, o = i & 127;
    float a = fc1b[o];
    const float* w = fc1w + o * HID; const float* p = P + r * HID;
    for (int k = 0; k < HID; ++k) a += p[k] * w[k];
    float inv = g1[o] * rsqrtf(v1[o] + EPS);
    a = a * inv + (b1[o] - m1[o] * inv);
    Z1[i] = 0.5f * a * (1.f + erff(a * 0.70710678118f));
  }
  __syncthreads();
  for (int i = tid; i < BSZ * 64; i += 256) {
    int r = i >> 6, o = i & 63;
    float a = fc2b[o];
    const float* w = fc2w + o * 128; const float* z = Z1 + r * 128;
    for (int k = 0; k < 128; ++k) a += z[k] * w[k];
    float inv = g2[o] * rsqrtf(v2[o] + EPS);
    a = a * inv + (b2[o] - m2[o] * inv);
    Z2[i] = 0.5f * a * (1.f + erff(a * 0.70710678118f));
  }
  __syncthreads();
  for (int i = tid; i < BSZ * 4; i += 256) {
    int r = i >> 2, o = i & 3;
    float a = fc3b[o];
    const float* w = fc3w + o * 64; const float* z = Z2 + r * 64;
    for (int k = 0; k < 64; ++k) a += z[k] * w[k];
    out[i] = a;
  }
}

// ---------------------------------------------------------------------------
extern "C" void kernel_launch(void* const* d_in, const int* in_sizes, int n_in,
                              void* d_out, int out_size, void* d_ws, size_t ws_size,
                              hipStream_t stream) {
  const float* F[64];
  for (int i = 0; i < n_in && i < 64; ++i) F[i] = (const float*)d_in[i];

  // input index maps for the two plausible flattening orders
  const float *x, *W3[3], *B3[3], *W5[3], *B5[3], *W7[3], *B7[3], *RW[3], *RB[3];
  const float *TG[3], *TB[3], *TM[3], *TV[3];
  const float *bng, *bnb, *bnm, *bnv, *lifw, *attw, *attb;
  const float *fc1w, *fc1b, *g1, *b1, *m1, *v1;
  const float *fc2w, *fc2b, *g2, *b2, *m2, *v2, *fc3w, *fc3b;

  if (in_sizes[0] == BSZ * 264 * TLEN) {
    // insertion order: x first, nested dicts in insertion order
    x = F[0];
    for (int l = 0; l < 3; ++l) {
      int o = 1 + l * 12;
      W3[l]=F[o+0]; B3[l]=F[o+1]; W5[l]=F[o+2]; B5[l]=F[o+3]; W7[l]=F[o+4]; B7[l]=F[o+5];
      RW[l]=F[o+6]; RB[l]=F[o+7]; TG[l]=F[o+8]; TB[l]=F[o+9]; TM[l]=F[o+10]; TV[l]=F[o+11];
    }
    bng=F[37]; bnb=F[38]; bnm=F[39]; bnv=F[40]; lifw=F[41];
    attw=F[42]; attb=F[43];
    fc1w=F[44]; fc1b=F[45]; g1=F[46]; b1=F[47]; m1=F[48]; v1=F[49];
    fc2w=F[50]; fc2b=F[51]; g2=F[52]; b2=F[53]; m2=F[54]; v2=F[55];
    fc3w=F[56]; fc3b=F[57];
  } else {
    // jax tree_leaves order: dict keys sorted alphabetically, params before x
    attb=F[0]; attw=F[1];
    b1=F[2]; g1=F[3]; m1=F[4]; v1=F[5];
    b2=F[6]; g2=F[7]; m2=F[8]; v2=F[9];
    bnb=F[10]; bng=F[11]; bnm=F[12]; bnv=F[13];
    fc1b=F[14]; fc1w=F[15]; fc2b=F[16]; fc2w=F[17]; fc3b=F[18]; fc3w=F[19];
    lifw=F[20];
    for (int l = 0; l < 3; ++l) {
      int o = 21 + l * 12;
      B3[l]=F[o+0]; B5[l]=F[o+1]; B7[l]=F[o+2];
      TB[l]=F[o+3]; TG[l]=F[o+4]; TM[l]=F[o+5]; TV[l]=F[o+6];
      RB[l]=F[o+7]; RW[l]=F[o+8]; W3[l]=F[o+9]; W5[l]=F[o+10]; W7[l]=F[o+11];
    }
    x = F[57];
  }

  // workspace carve (all offsets 64B aligned)
  char* ws = (char*)d_ws;
  const size_t ACT = (size_t)BSZ * HID * TLEN * sizeof(bf16); // 50,331,648
  bf16*  H1   = (bf16*)(ws + 0);
  bf16*  H2   = (bf16*)(ws + ACT);
  bf16*  H3   = (bf16*)(ws + 2 * ACT);
  bf16*  SPK  = H1;                                  // alias: H1 dead after conv2
  size_t off  = 3 * ACT;
  bf16*  WP1  = (bf16*)(ws + off); off += (size_t)8 * HID * 288 * sizeof(bf16);
  bf16*  WP2  = (bf16*)(ws + off); off += (size_t)8 * HID * 192 * sizeof(bf16);
  bf16*  WP3  = (bf16*)(ws + off); off += (size_t)8 * HID * 192 * sizeof(bf16);
  float* PRM1 = (float*)(ws + off); off += 3 * HID * sizeof(float) + 64;
  float* PRM2 = (float*)(ws + off); off += 3 * HID * sizeof(float) + 64;
  float* PRM3 = (float*)(ws + off); off += 3 * HID * sizeof(float) + 64;
  float* LG   = (float*)(ws + off); off += (size_t)BSZ * TLEN * sizeof(float);
  float* WSF  = (float*)(ws + off); off += (size_t)BSZ * TLEN * sizeof(float);
  float* POOL = (float*)(ws + off); off += (size_t)BSZ * HID * sizeof(float);
  (void)ws_size;

  // pack weights + fold BN (layer1 Cin=264 padded to 288; layers 2/3 Cin=192)
  pack_tcn<<<(8*HID*288 + 255)/256, 256, 0, stream>>>(
      W3[0],B3[0],W5[0],B5[0],W7[0],B7[0],RW[0],RB[0],TG[0],TB[0],TM[0],TV[0], WP1, PRM1, 264, 288);
  pack_tcn<<<(8*HID*192 + 255)/256, 256, 0, stream>>>(
      W3[1],B3[1],W5[1],B5[1],W7[1],B7[1],RW[1],RB[1],TG[1],TB[1],TM[1],TV[1], WP2, PRM2, 192, 192);
  pack_tcn<<<(8*HID*192 + 255)/256, 256, 0, stream>>>(
      W3[2],B3[2],W5[2],B5[2],W7[2],B7[2],RW[2],RB[2],TG[2],TB[2],TM[2],TV[2], WP3, PRM3, 192, 192);

  // WMMA conv blocks: grid = 32 batches * 64 time-tiles
  dim3 cg(BSZ * 64);
  tcn_conv<float><<<cg, 256, 0, stream>>>(x,  WP1, PRM1, H1, 264, 288, 1);
  tcn_conv<bf16 ><<<cg, 256, 0, stream>>>(H1, WP2, PRM2, H2, 192, 192, 2);
  tcn_conv<bf16 ><<<cg, 256, 0, stream>>>(H2, WP3, PRM3, H3, 192, 192, 4);

  // BN + LIF scan -> spikes
  lif_scan<<<BSZ, HID, 0, stream>>>(H3, bng, bnb, bnm, bnv, lifw, SPK);

  // attention pooling
  attn_logits<<<dim3(TLEN/256, BSZ), 256, 0, stream>>>(SPK, attw, attb, LG);
  softmax_t<<<BSZ, 256, 0, stream>>>(LG, WSF);
  attn_pool<<<BSZ, HID, 0, stream>>>(SPK, WSF, POOL);

  // classifier
  classifier<<<1, 256, 0, stream>>>(POOL,
      fc1w, fc1b, g1, b1, m1, v1,
      fc2w, fc2b, g2, b2, m2, v2,
      fc3w, fc3b, (float*)d_out);
  (void)out_size; (void)n_in;
}
